// CDA_78305843741382
// MI455X (gfx1250) — matrix operand order repaired and checked by reading, not compiled, not run on hardware
//
#include <hip/hip_runtime.h>

typedef unsigned short u16;
typedef __attribute__((ext_vector_type(16))) __bf16 v16bf;
typedef __attribute__((ext_vector_type(8)))  float  v8f;

#define BATCH 4
#define NTOK  16384
#define DIM   256
#define HEADS 2
#define CH    128   // DIM / HEADS

// ---------- bf16 helpers (manual, RNE) ----------
__device__ __forceinline__ float bf2f(u16 u) {
    union { unsigned int i; float f; } c; c.i = ((unsigned int)u) << 16; return c.f;
}
__device__ __forceinline__ u16 f2bf(float f) {
    union { float f; unsigned int i; } c; c.f = f;
    unsigned int x = c.i;
    return (u16)((x + 0x7fffu + ((x >> 16) & 1u)) >> 16);
}

// ---------- WMMA fragment loaders (wave32, 16x16x32 bf16) ----------
// All loads are expressed as ONE base pointer + compile-time element offset so the
// backend folds them into the 24-bit global / 16-bit DS instruction offsets
// (one address register pair per loop instead of eight).
__device__ __forceinline__ v16bf frag_a(const u16* base, int off, int half) {
    union { v16bf v; uint4 q[2]; } u;
    u.q[0] = *(const uint4*)(base + off + 8 * half);
    u.q[1] = *(const uint4*)(base + off + 16 + 8 * half);
    return u.v;
}
__device__ __forceinline__ v16bf frag_b(const u16* base, int off, int half) {
    union { v16bf v; uint4 q[2]; } u;
    const u16* p = base + off + 8 * half;
    u.q[0] = *(const uint4*)(p);
    u.q[1] = *(const uint4*)(p + 16);
    return u.v;
}
__device__ __forceinline__ v8f wmma_bf16(v16bf a, v16bf b, v8f c) {
    return __builtin_amdgcn_wmma_f32_16x16x32_bf16(false, a, false, b, (short)0, c, false, false);
}

// ---------- Tensor Data Mover (TDM) support, guarded per-toolchain ----------
#if __has_builtin(__builtin_amdgcn_tensor_load_to_lds) && __has_builtin(__builtin_amdgcn_s_wait_tensorcnt)
#define HAVE_TDM 1
#else
#define HAVE_TDM 0
#endif

#if HAVE_TDM
typedef unsigned int tdm_v4u __attribute__((ext_vector_type(4)));
typedef int          tdm_v8i __attribute__((ext_vector_type(8)));
typedef int          tdm_v4i __attribute__((ext_vector_type(4)));

__device__ __forceinline__ unsigned lds_off(void* p) {
    // generic -> LDS(AS3) pointer, then ptrtoint gives the LDS byte address
    return (unsigned)(unsigned long long)(__attribute__((address_space(3))) void*)p;
}

// 2-D tile DMA: tile_y rows x tile_x elements of 2 bytes, row stride = stride_elts.
// D# layout per CDNA5 ISA 8.3 (group0) / 8.4 (group1); groups 2-3 zero (<=2D).
__device__ __forceinline__ void tdm_load_2d(unsigned long long gaddr, unsigned lds_addr,
                                            unsigned rowlen_elts, unsigned rows,
                                            unsigned tile_x, unsigned tile_y,
                                            unsigned stride_elts) {
    tdm_v4u g0;
    g0[0] = 1u;                                               // count=1, user mode
    g0[1] = lds_addr;                                         // LDS byte address
    g0[2] = (unsigned)(gaddr & 0xffffffffull);                // global_addr[31:0]
    g0[3] = (unsigned)((gaddr >> 32) & 0x1ffffffull) | (2u << 30);  // [56:32] | type=2
    tdm_v8i g1;
    g1[0] = (int)(1u << 16);                                  // mask=0, data_size=2B
    g1[1] = (int)((rowlen_elts & 0xffffu) << 16);             // tensor_dim0[15:0]
    g1[2] = (int)((rowlen_elts >> 16) & 0xffffu) | (int)((rows & 0xffffu) << 16);
    g1[3] = (int)((rows >> 16) & 0xffffu) | (int)((tile_x & 0xffffu) << 16);
    g1[4] = (int)(tile_y & 0xffffu);                          // tile_dim1 | tile_dim2=0
    g1[5] = (int)stride_elts;                                 // tensor_dim0_stride[31:0]
    g1[6] = 0;
    g1[7] = 0;
    tdm_v4i z4 = {0, 0, 0, 0};
#if __clang_major__ >= 23
    tdm_v8i z8 = {0, 0, 0, 0, 0, 0, 0, 0};
    __builtin_amdgcn_tensor_load_to_lds(g0, g1, z4, z4, z8, 0);
#else
    __builtin_amdgcn_tensor_load_to_lds(g0, g1, z4, z4, 0);
#endif
}
#endif  // HAVE_TDM

// ---------- K0: f32 -> bf16 weight conversion ----------
__global__ void cvt_kernel(const float* __restrict__ s, u16* __restrict__ d, int n) {
    int i = blockIdx.x * 256 + threadIdx.x;
    if (i < n) d[i] = f2bf(s[i]);
}

// ---------- K1: LayerNorm (one wave per token, both tensors) ----------
__global__ void ln_kernel(const float* __restrict__ inR, const float* __restrict__ inS,
                          const float* __restrict__ lnR_w, const float* __restrict__ lnR_b,
                          const float* __restrict__ lnS_w, const float* __restrict__ lnS_b,
                          u16* __restrict__ xn, u16* __restrict__ yn) {
    int gw = blockIdx.x * 8 + (threadIdx.x >> 5);
    int lane = threadIdx.x & 31;
    const int total = BATCH * NTOK;
    const float *src, *w, *bb; u16* dst; int t;
    if (gw < total) { t = gw;        src = inR; w = lnR_w; bb = lnR_b; dst = xn; }
    else            { t = gw - total; src = inS; w = lnS_w; bb = lnS_b; dst = yn; }
    const float4* p = (const float4*)(src + (size_t)t * DIM) + lane * 2;
    float4 a = p[0], b2 = p[1];
    float v[8] = { a.x, a.y, a.z, a.w, b2.x, b2.y, b2.z, b2.w };
    float s = 0.f, ss = 0.f;
#pragma unroll
    for (int i = 0; i < 8; i++) { s += v[i]; ss += v[i] * v[i]; }
#pragma unroll
    for (int off = 16; off > 0; off >>= 1) {
        s  += __shfl_xor(s,  off, 32);
        ss += __shfl_xor(ss, off, 32);
    }
    float mu   = s * (1.f / DIM);
    float var  = ss * (1.f / DIM) - mu * mu;
    float rstd = rsqrtf(var + 1e-5f);
    uint4 outv; unsigned* po = (unsigned*)&outv;
#pragma unroll
    for (int i = 0; i < 4; i++) {
        int d0 = lane * 8 + 2 * i;
        float lo = (v[2 * i]     - mu) * rstd * w[d0]     + bb[d0];
        float hi = (v[2 * i + 1] - mu) * rstd * w[d0 + 1] + bb[d0 + 1];
        po[i] = (unsigned)f2bf(lo) | ((unsigned)f2bf(hi) << 16);
    }
    *((uint4*)(dst + (size_t)t * DIM) + lane) = outv;
}

// ---------- K2: fused Q / KV projection GEMM ----------
// grid.x: token tiles of 128; grid.y: 6 strips of 128 outputs
// B token-tile (shared by all 8 waves) staged once per workgroup via TDM.
__global__ void proj_kernel(const u16* __restrict__ yn, const u16* __restrict__ xn,
                            const u16* __restrict__ Wq, const u16* __restrict__ Wkv,
                            const float* __restrict__ bq, const float* __restrict__ bkv,
                            u16* __restrict__ q_t, u16* __restrict__ k_t,
                            u16* __restrict__ v_tok) {
    int wave = threadIdx.x >> 5, lane = threadIdx.x & 31;
    int half = lane >> 4, lm = lane & 15;
    int g0 = blockIdx.x * 128;       // global token base (b*N + n)
    int strip = blockIdx.y;
    const u16 *src, *W; const float* bias; int o0;
    if (strip < 2) { src = yn; W = Wq;  bias = bq;  o0 = strip * 128 + wave * 16; }
    else           { src = xn; W = Wkv; bias = bkv; o0 = (strip - 2) * 128 + wave * 16; }

#if HAVE_TDM
    __shared__ u16 ldsb[128 * DIM];  // 128 tokens x 256 ch = 64KB
    if (threadIdx.x < 32) {
        tdm_load_2d((unsigned long long)(size_t)(src + (size_t)g0 * DIM),
                    lds_off(ldsb), DIM, 128, DIM, 128, DIM);
        __builtin_amdgcn_s_wait_tensorcnt(0);
    }
    __syncthreads();
    const u16* bbase = ldsb + lm * DIM;
#else
    const u16* bbase = src + (size_t)(g0 + lm) * DIM;
#endif
    const u16* arow = W + (size_t)(o0 + lm) * DIM;
    v8f acc[8] = {};
#pragma unroll
    for (int k0 = 0; k0 < DIM; k0 += 32) {
        v16bf a = frag_a(arow, k0, half);
#pragma unroll
        for (int j = 0; j < 8; j++)
            acc[j] = wmma_bf16(a, frag_b(bbase, j * 16 * DIM + k0, half), acc[j]);
    }
    int b  = g0 >> 14;
    int nb = g0 & (NTOK - 1);
    if (strip < 4) {  // q or k: channel-major [b][ch][n]; one base + const offsets
        u16* dst = (strip < 2) ? q_t : k_t;
        u16* sbase = dst + ((size_t)(b * DIM + o0 + 8 * half)) * NTOK + nb + lm;
#pragma unroll
        for (int j = 0; j < 8; j++)
#pragma unroll
            for (int r = 0; r < 8; r++)
                sbase[r * NTOK + j * 16] = f2bf(acc[j][r] + bias[o0 + 8 * half + r]);
    } else {          // v: token-major, 8 contiguous bf16 per tile -> packed 16B store
        int dv0 = (strip - 4) * 128 + wave * 16 + 8 * half;
        u16* vbase = v_tok + (size_t)(g0 + lm) * DIM + dv0;
#pragma unroll
        for (int j = 0; j < 8; j++) {
            uint4 outv; unsigned* po = (unsigned*)&outv;
#pragma unroll
            for (int i = 0; i < 4; i++) {
                float lo = acc[j][2 * i]     + bias[256 + dv0 + 2 * i];
                float hi = acc[j][2 * i + 1] + bias[256 + dv0 + 2 * i + 1];
                po[i] = (unsigned)f2bf(lo) | ((unsigned)f2bf(hi) << 16);
            }
            *(uint4*)(vbase + j * 16 * DIM) = outv;
        }
    }
}

// ---------- K3: per-channel inverse L2 norms of q / k over n ----------
__global__ void rownorm_kernel(const u16* __restrict__ q_t, const u16* __restrict__ k_t,
                               float* __restrict__ qinv, float* __restrict__ kinv) {
    __shared__ float red[8];
    const u16* src = blockIdx.y ? k_t : q_t;
    float* dst = blockIdx.y ? kinv : qinv;
    const uint4* row = (const uint4*)(src + (size_t)blockIdx.x * NTOK);
    float s = 0.f;
    for (int i = threadIdx.x; i < NTOK / 8; i += 256) {
        uint4 q = row[i];
        unsigned* pu = (unsigned*)&q;
#pragma unroll
        for (int w2 = 0; w2 < 4; w2++) {
            float lo = bf2f((u16)(pu[w2] & 0xffffu));
            float hi = bf2f((u16)(pu[w2] >> 16));
            s += lo * lo + hi * hi;
        }
    }
#pragma unroll
    for (int off = 16; off > 0; off >>= 1) s += __shfl_xor(s, off, 32);
    if ((threadIdx.x & 31) == 0) red[threadIdx.x >> 5] = s;
    __syncthreads();
    if (threadIdx.x == 0) {
        float t = 0.f;
#pragma unroll
        for (int i = 0; i < 8; i++) t += red[i];
        dst[blockIdx.x] = 1.f / fmaxf(sqrtf(t), 1e-12f);
    }
}

// ---------- K4: gram = q . k^T over n ----------
// K tiles (shared by all 8 waves) double-buffer through LDS via the TDM on
// TENSORcnt; q rows (per-wave exclusive) stream from global with prefetch.
__global__ void gram_kernel(const u16* __restrict__ q_t, const u16* __restrict__ k_t,
                            float* __restrict__ gram) {
    int wave = threadIdx.x >> 5, lane = threadIdx.x & 31;
    int half = lane >> 4, lm = lane & 15;
    int bh = blockIdx.y, b = bh >> 1, h = bh & 1;
    int n0c = blockIdx.x * 1024;
    const u16* qb = q_t + ((size_t)(b * DIM + h * CH)) * NTOK;
    const u16* kb = k_t + ((size_t)(b * DIM + h * CH)) * NTOK;
    int c0 = wave * 16;
    const u16* arow = qb + (size_t)(c0 + lm) * NTOK;
    v8f acc[8] = {};

#if HAVE_TDM
    __shared__ u16 ldsk[2][CH * 128];   // double-buffered 128ch x 128tok K tile (2x32KB)
    unsigned lbase = lds_off(&ldsk[0][0]);
    const int NCHK = 1024 / 128;        // 8 chunks per workgroup
    unsigned long long kaddr = (unsigned long long)(size_t)kb;
    if (threadIdx.x < 32) {             // wave0 drives the TDM
        tdm_load_2d(kaddr + (size_t)n0c * 2, lbase, NTOK, CH, 128, CH, NTOK);
    }
    for (int ci = 0; ci < NCHK; ci++) {
        if (threadIdx.x < 32) {
            if (ci + 1 < NCHK) {        // issue next chunk, then wait for current
                tdm_load_2d(kaddr + (size_t)(n0c + (ci + 1) * 128) * 2,
                            lbase + (unsigned)(((ci + 1) & 1) * (CH * 128 * 2)),
                            NTOK, CH, 128, CH, NTOK);
                __builtin_amdgcn_s_wait_tensorcnt(1);
            } else {
                __builtin_amdgcn_s_wait_tensorcnt(0);
            }
        }
        __syncthreads();                // publish DMA'd tile to all waves
        const u16* bbase = &ldsk[ci & 1][0] + lm * 128;
        int nbase = n0c + ci * 128;
        __builtin_prefetch(arow + nbase + 128, 0, 3);   // next q chunk
#pragma unroll
        for (int kk = 0; kk < 128; kk += 32) {
            v16bf a = frag_a(arow, nbase + kk, half);
#pragma unroll
            for (int j = 0; j < 8; j++)
                acc[j] = wmma_bf16(a, frag_b(bbase, j * 16 * 128 + kk, half), acc[j]);
        }
        __syncthreads();                // all waves done with this buffer
    }
#else
    const u16* bbase = kb + (size_t)lm * NTOK;
    for (int n0 = n0c; n0 < n0c + 1024; n0 += 32) {
        v16bf a = frag_a(arow, n0, half);
#pragma unroll
        for (int j = 0; j < 8; j++)
            acc[j] = wmma_bf16(a, frag_b(bbase, j * 16 * NTOK + n0, half), acc[j]);
    }
#endif

    float* gbase = gram + (size_t)bh * CH * CH + (c0 + 8 * half) * CH + lm;
#pragma unroll
    for (int j = 0; j < 8; j++)
#pragma unroll
        for (int r = 0; r < 8; r++)
            atomicAdd(&gbase[r * CH + j * 16], acc[j][r]);
}

// ---------- K5: scale by qinv*kinv*temperature, row softmax, -> bf16 ----------
__global__ void softmax_kernel(const float* __restrict__ gram,
                               const float* __restrict__ qinv, const float* __restrict__ kinv,
                               const float* __restrict__ temperature,
                               u16* __restrict__ attn) {
    int wave = threadIdx.x >> 5, lane = threadIdx.x & 31;
    int row = blockIdx.x * 8 + wave;   // 0..1023
    int bh = row >> 7, c = row & 127;
    int b = bh >> 1, h = bh & 1;
    const float* g = gram + (size_t)bh * CH * CH + c * CH;
    float qi = qinv[b * DIM + h * CH + c] * temperature[h];
    float vals[4];
#pragma unroll
    for (int t = 0; t < 4; t++) {
        int d = lane * 4 + t;
        vals[t] = g[d] * qi * kinv[b * DIM + h * CH + d];
    }
    float m = fmaxf(fmaxf(vals[0], vals[1]), fmaxf(vals[2], vals[3]));
#pragma unroll
    for (int off = 16; off > 0; off >>= 1) m = fmaxf(m, __shfl_xor(m, off, 32));
    float s = 0.f;
#pragma unroll
    for (int t = 0; t < 4; t++) { vals[t] = __expf(vals[t] - m); s += vals[t]; }
#pragma unroll
    for (int off = 16; off > 0; off >>= 1) s += __shfl_xor(s, off, 32);
    float inv = 1.f / s;
    unsigned lo = (unsigned)f2bf(vals[0] * inv) | ((unsigned)f2bf(vals[1] * inv) << 16);
    unsigned hi = (unsigned)f2bf(vals[2] * inv) | ((unsigned)f2bf(vals[3] * inv) << 16);
    *(uint2*)(attn + (size_t)bh * CH * CH + c * CH + lane * 4) = make_uint2(lo, hi);
}

// ---------- K6: out1 = attn @ v, token-major bf16 output ----------
// V token-tile (shared by all 8 waves) staged once per workgroup via TDM.
__global__ void attnv_kernel(const u16* __restrict__ attn, const u16* __restrict__ v_tok,
                             u16* __restrict__ o1) {
    int wave = threadIdx.x >> 5, lane = threadIdx.x & 31;
    int half = lane >> 4, lm = lane & 15;
    int g0 = blockIdx.x * 128;
    int b = g0 >> 14;
    int h = blockIdx.y;
    const u16* A = attn + (size_t)(b * HEADS + h) * CH * CH;
    int c0 = wave * 16;
    const u16* arow = A + (size_t)(c0 + lm) * CH;

#if HAVE_TDM
    __shared__ u16 ldsv[128 * CH];   // 128 tokens x 128 head-ch = 32KB
    if (threadIdx.x < 32) {
        tdm_load_2d((unsigned long long)(size_t)(v_tok + (size_t)g0 * DIM + h * CH),
                    lds_off(ldsv), CH, 128, CH, 128, DIM);
        __builtin_amdgcn_s_wait_tensorcnt(0);
    }
    __syncthreads();
    const u16* bbase = ldsv + lm * CH;
    const int bstride = CH;
#else
    const u16* bbase = v_tok + (size_t)(g0 + lm) * DIM + h * CH;
    const int bstride = DIM;
#endif
    v8f acc[8] = {};
#pragma unroll
    for (int k0 = 0; k0 < CH; k0 += 32) {
        v16bf a = frag_a(arow, k0, half);
#pragma unroll
        for (int j = 0; j < 8; j++)
            acc[j] = wmma_bf16(a, frag_b(bbase, j * 16 * bstride + k0, half), acc[j]);
    }
    u16* obase = o1 + (size_t)(g0 + lm) * DIM + h * CH + c0 + 8 * half;
#pragma unroll
    for (int j = 0; j < 8; j++) {
        uint4 outv; unsigned* po = (unsigned*)&outv;
#pragma unroll
        for (int i = 0; i < 4; i++)
            po[i] = (unsigned)f2bf(acc[j][2 * i]) | ((unsigned)f2bf(acc[j][2 * i + 1]) << 16);
        *(uint4*)(obase + j * 16 * DIM) = outv;
    }
}

// ---------- K7: final projection, f32 output [b][n][o] ----------
__global__ void outproj_kernel(const u16* __restrict__ o1, const u16* __restrict__ Wo,
                               const float* __restrict__ bo, float* __restrict__ out) {
    int wave = threadIdx.x >> 5, lane = threadIdx.x & 31;
    int half = lane >> 4, lm = lane & 15;
    int g0 = blockIdx.x * 128;
    int o0 = blockIdx.y * 128 + wave * 16;

#if HAVE_TDM
    __shared__ u16 ldsb[128 * DIM];  // 128 tokens x 256 ch = 64KB
    if (threadIdx.x < 32) {
        tdm_load_2d((unsigned long long)(size_t)(o1 + (size_t)g0 * DIM),
                    lds_off(ldsb), DIM, 128, DIM, 128, DIM);
        __builtin_amdgcn_s_wait_tensorcnt(0);
    }
    __syncthreads();
    const u16* bbase = ldsb + lm * DIM;
#else
    const u16* bbase = o1 + (size_t)(g0 + lm) * DIM;
#endif
    const u16* arow = Wo + (size_t)(o0 + lm) * DIM;
    v8f acc[8] = {};
#pragma unroll
    for (int k0 = 0; k0 < DIM; k0 += 32) {
        v16bf a = frag_a(arow, k0, half);
#pragma unroll
        for (int j = 0; j < 8; j++)
            acc[j] = wmma_bf16(a, frag_b(bbase, j * 16 * DIM + k0, half), acc[j]);
    }
    float* fbase = out + (size_t)(g0 + lm) * DIM + o0 + 8 * half;
#pragma unroll
    for (int j = 0; j < 8; j++) {
        float4 lo4, hi4;
        lo4.x = acc[j][0] + bo[o0 + 8 * half + 0];
        lo4.y = acc[j][1] + bo[o0 + 8 * half + 1];
        lo4.z = acc[j][2] + bo[o0 + 8 * half + 2];
        lo4.w = acc[j][3] + bo[o0 + 8 * half + 3];
        hi4.x = acc[j][4] + bo[o0 + 8 * half + 4];
        hi4.y = acc[j][5] + bo[o0 + 8 * half + 5];
        hi4.z = acc[j][6] + bo[o0 + 8 * half + 6];
        hi4.w = acc[j][7] + bo[o0 + 8 * half + 7];
        *(float4*)(fbase + j * 16 * DIM)     = lo4;
        *(float4*)(fbase + j * 16 * DIM + 4) = hi4;
    }
}

extern "C" void kernel_launch(void* const* d_in, const int* in_sizes, int n_in,
                              void* d_out, int out_size, void* d_ws, size_t ws_size,
                              hipStream_t stream) {
    const float* input_R     = (const float*)d_in[0];
    const float* input_S     = (const float*)d_in[1];
    const float* lnS_w       = (const float*)d_in[2];
    const float* lnS_b       = (const float*)d_in[3];
    const float* lnR_w       = (const float*)d_in[4];
    const float* lnR_b       = (const float*)d_in[5];
    const float* Wq          = (const float*)d_in[6];
    const float* bq          = (const float*)d_in[7];
    const float* Wkv         = (const float*)d_in[8];
    const float* bkv         = (const float*)d_in[9];
    const float* Wo          = (const float*)d_in[10];
    const float* bo          = (const float*)d_in[11];
    const float* temperature = (const float*)d_in[12];

    char* ws = (char*)d_ws;
    const size_t S2 = (size_t)BATCH * NTOK * DIM * 2;  // one bf16 tensor: 33.5 MB
    u16* xn    = (u16*)(ws + 0 * S2);
    u16* yn    = (u16*)(ws + 1 * S2);
    u16* q_t   = (u16*)(ws + 2 * S2);
    u16* k_t   = (u16*)(ws + 3 * S2);
    u16* v_tok = (u16*)(ws + 4 * S2);
    char* p = ws + 5 * S2;
    float* gram = (float*)p;  p += (size_t)BATCH * HEADS * CH * CH * 4;
    u16* attn   = (u16*)p;    p += (size_t)BATCH * HEADS * CH * CH * 2;
    float* qinv = (float*)p;  p += (size_t)BATCH * DIM * 4;
    float* kinv = (float*)p;  p += (size_t)BATCH * DIM * 4;
    u16* Wq_bf  = (u16*)p;    p += (size_t)DIM * DIM * 2;
    u16* Wkv_bf = (u16*)p;    p += (size_t)2 * DIM * DIM * 2;
    u16* Wo_bf  = (u16*)p;    p += (size_t)DIM * DIM * 2;
    u16* o1 = yn;  // yn is dead after proj_kernel; reuse for attn@v output

    cvt_kernel<<<DIM * DIM / 256, 256, 0, stream>>>(Wq, Wq_bf, DIM * DIM);
    cvt_kernel<<<2 * DIM * DIM / 256, 256, 0, stream>>>(Wkv, Wkv_bf, 2 * DIM * DIM);
    cvt_kernel<<<DIM * DIM / 256, 256, 0, stream>>>(Wo, Wo_bf, DIM * DIM);
    ln_kernel<<<2 * BATCH * NTOK / 8, 256, 0, stream>>>(input_R, input_S,
                                                        lnR_w, lnR_b, lnS_w, lnS_b, xn, yn);
    hipMemsetAsync(gram, 0, (size_t)BATCH * HEADS * CH * CH * 4, stream);
    proj_kernel<<<dim3(BATCH * NTOK / 128, 6), 256, 0, stream>>>(yn, xn, Wq_bf, Wkv_bf,
                                                                 bq, bkv, q_t, k_t, v_tok);
    rownorm_kernel<<<dim3(BATCH * DIM, 2), 256, 0, stream>>>(q_t, k_t, qinv, kinv);
    gram_kernel<<<dim3(NTOK / 1024, BATCH * HEADS), 256, 0, stream>>>(q_t, k_t, gram);
    softmax_kernel<<<BATCH * HEADS * CH / 8, 256, 0, stream>>>(gram, qinv, kinv,
                                                               temperature, attn);
    attnv_kernel<<<dim3(BATCH * NTOK / 128, HEADS), 256, 0, stream>>>(attn, v_tok, o1);
    outproj_kernel<<<dim3(BATCH * NTOK / 128, 2), 256, 0, stream>>>(o1, Wo_bf, bo,
                                                                    (float*)d_out);
}
